// Encoder_85959475462477
// MI455X (gfx1250) — compile-verified, matrix operand order checked by reading, not run on hardware
//
#include <hip/hip_runtime.h>

typedef __attribute__((ext_vector_type(16))) __bf16 v16bf;
typedef __attribute__((ext_vector_type(8)))  __bf16 v8bf;
typedef __attribute__((ext_vector_type(4)))  __bf16 v4bf;
typedef __attribute__((ext_vector_type(8)))  float  v8f;
typedef __attribute__((ext_vector_type(4)))  float  v4f;

#define NEG_SLOPE 0.01f

// ---------------------------------------------------------------- feature probe
#if defined(__gfx1250__) && __has_builtin(__builtin_amdgcn_global_load_async_to_lds_b128)
#define USE_ASYNC 1
#else
#define USE_ASYNC 0
#endif

union V16U { v16bf v; v8bf h[2]; };

#if USE_ASYNC
// builtin expects: (v4i AS(1)*, v4i AS(3)*, imm offset, imm cpol)
typedef int v4i __attribute__((vector_size(4 * sizeof(int))));
typedef __attribute__((address_space(1))) v4i* gb128_t;
typedef __attribute__((address_space(3))) v4i* lb128_t;

__device__ __forceinline__ gb128_t to_global(const void* p) {
    return reinterpret_cast<gb128_t>(reinterpret_cast<unsigned long long>(p));
}
// Generic LDS address: low 32 bits are the LDS offset (aperture model); the
// ptrtoint/inttoptr round-trip is folded to an addrspacecast by the backend.
__device__ __forceinline__ lb128_t to_lds(void* p) {
    return reinterpret_cast<lb128_t>(
        (unsigned int)reinterpret_cast<unsigned long long>(p));
}

__device__ __forceinline__ void wait_async_le4() {
#if __has_builtin(__builtin_amdgcn_s_wait_asynccnt)
    __builtin_amdgcn_s_wait_asynccnt(4);
#else
    asm volatile("s_wait_asynccnt 0x4" ::: "memory");
#endif
}
#endif

// ---------------------------------------------------------------- utilities

__global__ void zero_scales(unsigned int* p) {
    if (threadIdx.x < 4) p[threadIdx.x] = 0u;
}

__global__ void absmax_kernel(const float* __restrict__ w, int n,
                              unsigned int* __restrict__ out) {
    __shared__ float red[256];
    float m = 0.f;
    for (int i = blockIdx.x * blockDim.x + threadIdx.x; i < n;
         i += gridDim.x * blockDim.x)
        m = fmaxf(m, fabsf(w[i]));
    red[threadIdx.x] = m;
    __syncthreads();
    for (int s = 128; s > 0; s >>= 1) {
        if ((int)threadIdx.x < s)
            red[threadIdx.x] = fmaxf(red[threadIdx.x], red[threadIdx.x + s]);
        __syncthreads();
    }
    if (threadIdx.x == 0)
        atomicMax(out, __float_as_uint(red[0]));  // values >= 0: bit-compare ok
}

// Brevitas-style: scale = amax/n; w_q = clip(round(w/scale),-n,n)*scale  (bf16 out)
__global__ void quant_kernel(const float* __restrict__ w, int n,
                             const unsigned int* __restrict__ amax_bits,
                             float nlev, __bf16* __restrict__ out) {
    float amax  = __uint_as_float(*amax_bits);
    float scale = amax / nlev;
    float inv   = nlev / amax;
    for (int i = blockIdx.x * blockDim.x + threadIdx.x; i < n;
         i += gridDim.x * blockDim.x) {
        float q = rintf(w[i] * inv);           // round-half-even like jnp.round
        q = fminf(fmaxf(q, -nlev), nlev);
        out[i] = (__bf16)(q * scale);
    }
}

__global__ void cvt_f32_to_bf16(const float* __restrict__ in,
                                __bf16* __restrict__ out, int n4) {
    for (int i = blockIdx.x * blockDim.x + threadIdx.x; i < n4;
         i += gridDim.x * blockDim.x) {
        v4f f = ((const v4f*)in)[i];
        v4bf r;
#pragma unroll
        for (int j = 0; j < 4; ++j) r[j] = (__bf16)f[j];
        ((v4bf*)out)[i] = r;
    }
}

// ---------------------------------------------------------------- GEMM

// C[M,N] = A[M,K](bf16) * W[N,K](bf16)^T + bias, LeakyReLU epilogue.
// Block tile 128x128, BK=32. 256 threads = 8 waves (4 M-waves x 2 N-waves),
// each wave computes 32x64 via 2x4 wmma_f32_16x16x32_bf16 accumulators.
#define BM 128
#define BN 128
#define BK 32
#define LDT 40   // padded LDS row stride (elems): 80B, 16B-aligned, skews banks

#if USE_ASYNC
#define STAGES 3  // 60KB LDS, prefetch depth 2
#else
#define STAGES 2
#endif

__device__ __forceinline__ v16bf load_frag(const __bf16* base, int lane, int tile16) {
    // lanes 0-15: rows 0-15 K=0..15 ; lanes 16-31: rows 0-15 K=16..31
    const __bf16* p = base + (tile16 + (lane & 15)) * LDT + ((lane >> 4) << 4);
    V16U u;
    u.h[0] = *(const v8bf*)(p);
    u.h[1] = *(const v8bf*)(p + 8);
    return u.v;
}

template <bool OUT_F32>
__global__ __launch_bounds__(256)
void gemm_bias_lrelu(const __bf16* __restrict__ A,
                     const __bf16* __restrict__ W,
                     const float* __restrict__ bias,
                     void* __restrict__ Out,
                     int M, int N, int K) {
    __shared__ __bf16 As[STAGES][BM * LDT];
    __shared__ __bf16 Bs[STAGES][BN * LDT];

    const int tid  = threadIdx.x;
    const int lane = tid & 31;
    const int wave = tid >> 5;
    const int wm   = wave >> 1;   // 0..3 -> M offset wm*32
    const int wn   = wave & 1;    // 0..1 -> N offset wn*64

    const int m0 = blockIdx.y * BM;
    const int n0 = blockIdx.x * BN;

    v8f acc[2][4];
#pragma unroll
    for (int mt = 0; mt < 2; ++mt)
#pragma unroll
        for (int nt = 0; nt < 4; ++nt) {
            v8f z = {0.f, 0.f, 0.f, 0.f, 0.f, 0.f, 0.f, 0.f};
            acc[mt][nt] = z;
        }

    const int kts = K / BK;

#if USE_ASYNC
    // Each stage moves A(128x32) + B(128x32) bf16 = 16KB via async B128 ops:
    // each thread issues 2 chunks per matrix (4 async ops per thread/stage).
    auto issue = [&](int kt) {
        const int s = kt % STAGES;
        const size_t kbase = (size_t)kt * BK;
#pragma unroll
        for (int c = 0; c < 2; ++c) {
            const int id  = tid + (c << 8);       // chunk id 0..511
            const int row = id >> 2;              // 4 chunks of 8 elems per row
            const int kof = (id & 3) << 3;
            const __bf16* ga = A + (size_t)(m0 + row) * K + kbase + kof;
            const __bf16* gb = W + (size_t)(n0 + row) * K + kbase + kof;
            __bf16* la = &As[s][row * LDT + kof];
            __bf16* lb = &Bs[s][row * LDT + kof];
            __builtin_amdgcn_global_load_async_to_lds_b128(to_global(ga), to_lds(la), 0, 0);
            __builtin_amdgcn_global_load_async_to_lds_b128(to_global(gb), to_lds(lb), 0, 0);
        }
    };

    issue(0);
    if (1 < kts) issue(1);
#else
    // register-staged fallback: thread t handles row t>>1, k-chunk (t&1)*16
    const int srow = tid >> 1;
    const int skof = (tid & 1) << 4;
    v8bf ra0, ra1, rb0, rb1;
    auto load_regs = [&](int kt) {
        const size_t k = (size_t)kt * BK + skof;
        const __bf16* gp = A + (size_t)(m0 + srow) * K + k;
        ra0 = *(const v8bf*)(gp);
        ra1 = *(const v8bf*)(gp + 8);
        const __bf16* wp = W + (size_t)(n0 + srow) * K + k;
        rb0 = *(const v8bf*)(wp);
        rb1 = *(const v8bf*)(wp + 8);
    };
    auto store_regs = [&](int buf) {
        __bf16* as = &As[buf][0];
        __bf16* bs = &Bs[buf][0];
        *(v8bf*)(as + srow * LDT + skof)     = ra0;
        *(v8bf*)(as + srow * LDT + skof + 8) = ra1;
        *(v8bf*)(bs + srow * LDT + skof)     = rb0;
        *(v8bf*)(bs + srow * LDT + skof + 8) = rb1;
    };
    load_regs(0);
    store_regs(0);
    __syncthreads();
#endif

    for (int kt = 0; kt < kts; ++kt) {
        const int cur = kt % STAGES;
#if USE_ASYNC
        wait_async_le4();        // oldest stage (kt) has landed in LDS
        __syncthreads();         // ..for every wave; also fences stage kt-1 compute
        if (kt + 2 < kts) issue(kt + 2);   // overwrites buf (kt-1)%3: safe post-barrier
#else
        const bool more = (kt + 1) < kts;
        if (more) load_regs(kt + 1);
#endif

        const __bf16* a_base = &As[cur][0];
        const __bf16* b_base = &Bs[cur][0];
        v16bf afrag[2], bfrag[4];
#pragma unroll
        for (int mt = 0; mt < 2; ++mt)
            afrag[mt] = load_frag(a_base, lane, wm * 32 + mt * 16);
#pragma unroll
        for (int nt = 0; nt < 4; ++nt)
            bfrag[nt] = load_frag(b_base, lane, wn * 64 + nt * 16);
#pragma unroll
        for (int mt = 0; mt < 2; ++mt)
#pragma unroll
            for (int nt = 0; nt < 4; ++nt)
                acc[mt][nt] = __builtin_amdgcn_wmma_f32_16x16x32_bf16(
                    false, afrag[mt], false, bfrag[nt],
                    (short)0, acc[mt][nt], false, false);

#if !USE_ASYNC
        if (more) store_regs((kt + 1) & 1);
        __syncthreads();
#endif
    }

    // epilogue: bias + LeakyReLU
    // C/D layout: VGPR i -> row (lane>=16 ? 8 : 0)+i, col = lane&15
#pragma unroll
    for (int mt = 0; mt < 2; ++mt) {
#pragma unroll
        for (int nt = 0; nt < 4; ++nt) {
            const int ccol  = n0 + wn * 64 + nt * 16 + (lane & 15);
            const int crow0 = m0 + wm * 32 + mt * 16 + ((lane >> 4) << 3);
            const float bv  = bias[ccol];
#pragma unroll
            for (int i = 0; i < 8; ++i) {
                float v = acc[mt][nt][i] + bv;
                v = (v >= 0.f) ? v : NEG_SLOPE * v;
                const size_t idx = (size_t)(crow0 + i) * N + ccol;
                if constexpr (OUT_F32) ((float*)Out)[idx] = v;
                else                   ((__bf16*)Out)[idx] = (__bf16)v;
            }
        }
    }
}

// ---------------------------------------------------------------- launch

extern "C" void kernel_launch(void* const* d_in, const int* in_sizes, int n_in,
                              void* d_out, int out_size, void* d_ws, size_t ws_size,
                              hipStream_t stream) {
    (void)in_sizes; (void)n_in; (void)out_size; (void)ws_size;
    const float* x  = (const float*)d_in[0];
    const float* W0 = (const float*)d_in[1];
    const float* b0 = (const float*)d_in[2];
    const float* W1 = (const float*)d_in[3];
    const float* b1 = (const float*)d_in[4];
    const float* W2 = (const float*)d_in[5];
    const float* b2 = (const float*)d_in[6];

    const int Nrows = 32768, Din = 1024, Dout = 2048;

    // workspace layout (bf16 scratch)
    char* ws = (char*)d_ws;
    unsigned int* amax = (unsigned int*)ws;                 // 3 slots
    __bf16* wq0 = (__bf16*)(ws + 256);                      // 2048*1024
    __bf16* wq1 = wq0 + (size_t)Dout * Din;                 // 2048*2048
    __bf16* wq2 = wq1 + (size_t)Dout * Dout;                // 2048*2048
    __bf16* h2  = wq2 + (size_t)Dout * Dout;                // 32768*2048 bf16
    __bf16* xbf = h2  + (size_t)Nrows * Dout;               // 32768*1024 bf16
    __bf16* h1  = (__bf16*)d_out;   // alias: fully overwritten by final layer
    float*  out = (float*)d_out;

    zero_scales<<<1, 32, 0, stream>>>(amax);
    absmax_kernel<<<512, 256, 0, stream>>>(W0, Dout * Din,  amax + 0);
    absmax_kernel<<<512, 256, 0, stream>>>(W1, Dout * Dout, amax + 1);
    absmax_kernel<<<512, 256, 0, stream>>>(W2, Dout * Dout, amax + 2);
    quant_kernel<<<2048, 256, 0, stream>>>(W0, Dout * Din,  amax + 0, 127.f, wq0);
    quant_kernel<<<4096, 256, 0, stream>>>(W1, Dout * Dout, amax + 1, 7.f,   wq1);
    quant_kernel<<<4096, 256, 0, stream>>>(W2, Dout * Dout, amax + 2, 7.f,   wq2);
    cvt_f32_to_bf16<<<4096, 256, 0, stream>>>(x, xbf, (Nrows * Din) / 4);

    dim3 grid(Dout / BN, Nrows / BM);  // (16, 256)
    gemm_bias_lrelu<false><<<grid, 256, 0, stream>>>(xbf, wq0, b0, h1,  Nrows, Dout, Din);
    gemm_bias_lrelu<false><<<grid, 256, 0, stream>>>(h1,  wq1, b1, h2,  Nrows, Dout, Dout);
    gemm_bias_lrelu<true ><<<grid, 256, 0, stream>>>(h2,  wq2, b2, out, Nrows, Dout, Dout);
}
